// T_Mamba_27685359190709
// MI455X (gfx1250) — compile-verified, matrix operand order checked by reading, not run on hardware
//
#include <hip/hip_runtime.h>
#include <hip/hip_bf16.h>
#include <math.h>

typedef __attribute__((ext_vector_type(16))) _Float16 v16h;
typedef __attribute__((ext_vector_type(8)))  float    v8f;

#define WMMA_F16(A_,B_,C_) __builtin_amdgcn_wmma_f32_16x16x32_f16(false,(A_),false,(B_),(short)0,(C_),false,false)

static constexpr int Bn  = 32;
static constexpr int Lt  = 2048;   // L
static constexpr int LPt = 1024;   // L/2
static constexpr int NIN = 12;
static constexpr int DM  = 128;    // d_model
static constexpr int DI  = 256;    // d_inner
static constexpr int DSt = 256;    // d_state
static constexpr int DTR = 8;      // dt_rank
static constexpr int T16c = Lt / 16;    // conv-domain 16-token tiles
static constexpr int T16p = LPt / 16;   // pooled-domain 16-token tiles

// K index held by fragment half j on lane `lane` (wmma_*_16x16x32 16-bit layout, ISA 7.12.2)
__device__ __forceinline__ int kmap16(int lane, int j) {
  return (j & 7) + ((j >> 3) << 4) + ((lane >> 4) << 3);
}

// Element index inside a fragment-packed activation buffer.
// Layout: [b][kk/32][l/16][lane][j] ; lane = ((kk&31)>>3 & 1)*16 + (l&15) ; j = inverse kmap of kk&31.
__device__ __forceinline__ size_t pidx(int b, int kk, int l, int KS, int T16) {
  int rr = kk & 31;
  int jo = (rr & 7) | ((rr >> 4) << 3);
  int lo = (((rr >> 3) & 1) << 4) + (l & 15);
  return ((((size_t)b * KS + (kk >> 5)) * T16 + (l >> 4)) * 32 + lo) * 16 + jo;
}

// ---------------------------------------------------------------- prep kernels
// Pack input x (B,L,NIN) into fragment layout with channels padded 12->32 (zeros).
__global__ void k_pack_x(const float* __restrict__ x, _Float16* __restrict__ xp) {
  int i = blockIdx.x * 256 + threadIdx.x;
  if (i >= Bn * T16c * 512) return;
  int j = i & 15; int lane = (i >> 4) & 31; int tl = (i >> 9) % T16c; int b = (i >> 9) / T16c;
  int kk = kmap16(lane, j);
  int t = tl * 16 + (lane & 15);
  xp[i] = (kk < NIN) ? (_Float16)x[(b * Lt + t) * NIN + kk] : (_Float16)0.f;
}

// Pack weights [M][Cin][KW] (row-major f32) into per-(mtile,kslice,lane) v16h fragments.
// Packed K index = tap*CinP + ci, zero-filled for ci>=Cin or m>=Mreal.
__global__ void k_packw(const float* __restrict__ W, _Float16* __restrict__ P,
                        int Mreal, int Cin, int CinP, int KW, int slices, int total) {
  int i = blockIdx.x * 256 + threadIdx.x;
  if (i >= total) return;
  int j = i & 15; int lane = (i >> 4) & 31; int rest = i >> 9;
  int s = rest % slices; int mt = rest / slices;
  int m = mt * 16 + (lane & 15);
  int kkp = s * 32 + kmap16(lane, j);
  int tap = kkp / CinP, ci = kkp % CinP;
  _Float16 v = (_Float16)0.f;
  if (m < Mreal && ci < Cin) v = (_Float16)W[(m * Cin + ci) * KW + tap];
  P[i] = v;
}

__global__ void k_negexp(const float* __restrict__ alog, float* __restrict__ A, int n) {
  int i = blockIdx.x * 256 + threadIdx.x;
  if (i < n) A[i] = -__expf(alog[i]);
}

// ---------------------------------------------------------------- TCN conv as WMMA GEMM
// MODE 0: Y = relu(conv1 + b1)
// MODE 1: Y = relu( relu(conv1 + b1) + conv1x1(X2,W2) + b2 )
template <int CINP, int KW, int DIL, int COUT, int MODE, int CINP2>
__global__ void k_tcn(const v16h* __restrict__ Xp, const v16h* __restrict__ Wp,
                      const float* __restrict__ bias1,
                      const v16h* __restrict__ X2p, const v16h* __restrict__ W2p,
                      const float* __restrict__ bias2,
                      _Float16* __restrict__ Yp) {
  constexpr int KSa  = CINP / 32;
  constexpr int NSL  = KW * KSa;
  constexpr int KSa2 = CINP2 / 32;
  constexpr int KSo  = COUT / 32;
  int lane = threadIdx.x & 31;
  int wv   = threadIdx.x >> 5;                 // 4 waves, 16 tokens each
  int id = blockIdx.x;
  int tt4 = id % (Lt / 64); id /= (Lt / 64);
  int mt  = id % (COUT / 16);
  int b   = id / (COUT / 16);
  int t0 = tt4 * 64 + wv * 16;
  int tt = t0 >> 4;
  int n = lane & 15, hi = lane >> 4;
  v16h zz = {};

  v8f acc1 = {};
#pragma unroll
  for (int s = 0; s < NSL; ++s) {
    int tap = s / KSa, ks = s % KSa;
    int shift = (KW - 1 - tap) * DIL;
    v16h a = Wp[(mt * NSL + s) * 32 + lane];
    int ls = t0 + n - shift;
    int lsc = ls < 0 ? 0 : ls;
    v16h bf = Xp[(((size_t)b * KSa + ks) * T16c + (lsc >> 4)) * 32 + (hi * 16 + (lsc & 15))];
    if (ls < 0) bf = zz;
    acc1 = WMMA_F16(a, bf, acc1);
  }

  v8f acc2 = {};
  if constexpr (MODE == 1) {
#pragma unroll
    for (int ks = 0; ks < KSa2; ++ks) {
      v16h a  = W2p[(mt * KSa2 + ks) * 32 + lane];
      v16h bf = X2p[(((size_t)b * KSa2 + ks) * T16c + tt) * 32 + lane];
      acc2 = WMMA_F16(a, bf, acc2);
    }
  }

#pragma unroll
  for (int r = 0; r < 8; ++r) {
    int m = mt * 16 + r + hi * 8;
    float v = acc1[r] + bias1[m];
    if constexpr (MODE == 0) {
      v = fmaxf(v, 0.f);
    } else {
      v = fmaxf(v, 0.f) + acc2[r] + bias2[m];
      v = fmaxf(v, 0.f);
    }
    Yp[pidx(b, m, t0 + n, KSo, T16c)] = (_Float16)v;
  }
}

// ---------------------------------------------------------------- pool(2) + LayerNorm
__global__ void k_poolln(const _Float16* __restrict__ h2p, const float* __restrict__ g,
                         const float* __restrict__ be, const float* __restrict__ mask,
                         _Float16* __restrict__ lnp, float* __restrict__ outacc) {
  int b = blockIdx.x / LPt, lp = blockIdx.x % LPt;
  int c = threadIdx.x;  // 128 channels
  float v0 = (float)h2p[pidx(b, c, 2 * lp,     DM / 32, T16c)];
  float v1 = (float)h2p[pidx(b, c, 2 * lp + 1, DM / 32, T16c)];
  float v = fmaxf(v0, v1);
  __shared__ float red[128];
  red[c] = v; __syncthreads();
  for (int s = 64; s > 0; s >>= 1) { if (c < s) red[c] += red[c + s]; __syncthreads(); }
  float mu = red[0] / 128.f; __syncthreads();
  float dv = v - mu;
  red[c] = dv * dv; __syncthreads();
  for (int s = 64; s > 0; s >>= 1) { if (c < s) red[c] += red[c + s]; __syncthreads(); }
  float var = red[0] / 128.f;
  float o = dv * rsqrtf(var + 1e-5f) * g[c] + be[c];
  o *= mask[b * LPt + lp];
  lnp[pidx(b, c, lp, DM / 32, T16p)] = (_Float16)o;
  outacc[(b * LPt + lp) * DM + c] = 2.f * o;   // out_f + out_b each contribute ln once
}

// ---------------------------------------------------------------- Mamba in_proj (M=512,K=128)
__global__ void k_inproj(const v16h* __restrict__ lnp, const v16h* __restrict__ Wp,
                         _Float16* __restrict__ xzp, int flip) {
  int lane = threadIdx.x & 31, wv = threadIdx.x >> 5;
  int id = blockIdx.x;
  int tt4 = id % (LPt / 64); id /= (LPt / 64);
  int mt = id % 32; int b = id / 32;
  int t0 = tt4 * 64 + wv * 16;
  int tt = t0 >> 4;
  int n = lane & 15, hi = lane >> 4;
  int stile = flip ? (T16p - 1 - tt) : tt;
  int slane = hi * 16 + (flip ? (15 - n) : n);
  v8f acc = {};
#pragma unroll
  for (int ks = 0; ks < 4; ++ks) {
    v16h a  = Wp[(mt * 4 + ks) * 32 + lane];
    v16h bf = lnp[(((size_t)b * 4 + ks) * T16p + stile) * 32 + slane];
    acc = WMMA_F16(a, bf, acc);
  }
#pragma unroll
  for (int r = 0; r < 8; ++r) {
    int e = mt * 16 + r + hi * 8;
    xzp[pidx(b, e, t0 + n, 16, T16p)] = (_Float16)acc[r];
  }
}

// ---------------------------------------------------------------- depthwise conv(4) + silu
__global__ void k_dwconv(const _Float16* __restrict__ xzp, const float* __restrict__ cw,
                         const float* __restrict__ cb, _Float16* __restrict__ up,
                         _Float16* __restrict__ uth) {
  int i = blockIdx.x * 256 + threadIdx.x;
  if (i >= Bn * DI * LPt) return;
  int l = i % LPt; int e = (i / LPt) % DI; int b = i / (LPt * DI);
  float s = cb[e];
#pragma unroll
  for (int k = 0; k < 4; ++k) {
    int ts = l - 3 + k;
    if (ts >= 0) s += cw[e * 4 + k] * (float)xzp[pidx(b, e, ts, 16, T16p)];
  }
  float u = s / (1.f + __expf(-s));
  up[pidx(b, e, l, 8, T16p)] = (_Float16)u;      // fragment layout for x_proj B
  uth[(b * LPt + l) * DI + e] = (_Float16)u;     // token-major for scan/combine
}

// ---------------------------------------------------------------- x_proj (M=520->528,K=256)
__global__ void k_xproj(const v16h* __restrict__ up, const v16h* __restrict__ Wxp,
                        float* __restrict__ dtb, _Float16* __restrict__ Bch,
                        _Float16* __restrict__ Cch) {
  int lane = threadIdx.x & 31, wv = threadIdx.x >> 5;
  int id = blockIdx.x;
  int tt4 = id % (LPt / 64); id /= (LPt / 64);
  int mt = id % 33; int b = id / 33;
  int t0 = tt4 * 64 + wv * 16;
  int tt = t0 >> 4;
  int n = lane & 15, hi = lane >> 4;
  v8f acc = {};
#pragma unroll
  for (int ks = 0; ks < 8; ++ks) {
    v16h a  = Wxp[(mt * 8 + ks) * 32 + lane];
    v16h bf = up[(((size_t)b * 8 + ks) * T16p + tt) * 32 + lane];
    acc = WMMA_F16(a, bf, acc);
  }
#pragma unroll
  for (int r = 0; r < 8; ++r) {
    int g = mt * 16 + r + hi * 8;
    int l = t0 + n;
    float v = acc[r];
    if (g < DTR)                 dtb[(b * LPt + l) * DTR + g] = v;
    else if (g < DTR + DSt)      Bch[(b * LPt + l) * DSt + (g - DTR)] = (_Float16)v;
    else if (g < DTR + 2 * DSt)  Cch[(b * LPt + l) * DSt + (g - DTR - DSt)] = (_Float16)v;
  }
}

// ---------------------------------------------------------------- delta = softplus(dt@dt_w^T + dt_b)
__global__ void k_delta(const float* __restrict__ dtb, const float* __restrict__ dtw,
                        const float* __restrict__ dtbias, float* __restrict__ delta) {
  int i = blockIdx.x * 256 + threadIdx.x;
  if (i >= Bn * LPt * DI) return;
  int d = i % DI; int l = (i / DI) % LPt; int b = i / (DI * LPt);
  float s = dtbias[d];
#pragma unroll
  for (int r = 0; r < 8; ++r) s += dtb[(b * LPt + l) * DTR + r] * dtw[d * DTR + r];
  delta[(b * LPt + l) * DI + d] = (s > 20.f) ? s : log1pf(__expf(s));
}

// ---------------------------------------------------------------- selective scan
// grid = Bn*2 (batch x state-split). 1024 threads: d = tid>>2, q = tid&3, 32 states/thread in VGPRs.
// A is pre-scaled by log2(e) at load so the recurrence uses raw v_exp (exp2).
__global__ __launch_bounds__(1024) void k_scan(const float* __restrict__ Af,
                                               const float* __restrict__ delta,
                                               const _Float16* __restrict__ uth,
                                               const _Float16* __restrict__ Bch,
                                               const _Float16* __restrict__ Cch,
                                               float* __restrict__ ypart) {
  int b = blockIdx.x >> 1, split = blockIdx.x & 1;
  int tid = threadIdx.x;
  int d = tid >> 2, q = tid & 3;
  int sblk = split * 128;
  int sb = sblk + q * 32;
  const float LOG2E = 1.4426950408889634f;
  float hreg[32], Ar[32];
#pragma unroll
  for (int j = 0; j < 32; ++j) { hreg[j] = 0.f; Ar[j] = Af[d * DSt + sb + j] * LOG2E; }

  const int TC = 8;
  __shared__ __align__(16) _Float16 sB[TC * 128], sC[TC * 128], sU[TC * 256];
  __shared__ __align__(16) float sD[TC * 256];

  for (int c = 0; c < LPt / TC; ++c) {
    __syncthreads();
    // vectorized staging: uint4 (b128) copies
    for (int v = tid; v < 128; v += 1024) {            // sB,sC: 8 rows x 128 halves
      int t = c * TC + (v >> 4);
      int o = (v & 15) * 8;
      *(uint4*)&sB[(v >> 4) * 128 + o] =
          *(const uint4*)&Bch[((size_t)(b * LPt + t)) * DSt + sblk + o];
      *(uint4*)&sC[(v >> 4) * 128 + o] =
          *(const uint4*)&Cch[((size_t)(b * LPt + t)) * DSt + sblk + o];
    }
    for (int v = tid; v < 256; v += 1024) {            // sU: 8 rows x 256 halves
      int t = c * TC + (v >> 5);
      int o = (v & 31) * 8;
      *(uint4*)&sU[(v >> 5) * 256 + o] =
          *(const uint4*)&uth[((size_t)(b * LPt + t)) * DI + o];
    }
    for (int v = tid; v < 512; v += 1024) {            // sD: 8 rows x 256 floats
      int t = c * TC + (v >> 6);
      int o = (v & 63) * 4;
      *(uint4*)&sD[(v >> 6) * 256 + o] =
          *(const uint4*)&delta[((size_t)(b * LPt + t)) * DI + o];
    }
    if (c + 1 < LPt / TC) {
      __builtin_prefetch(&delta[((size_t)(b * LPt + (c + 1) * TC)) * DI + d], 0, 1);
    }
    __syncthreads();
#pragma unroll 1
    for (int s = 0; s < TC; ++s) {
      float dl = sD[s * 256 + d];
      float du = dl * (float)sU[s * 256 + d];
      float yp = 0.f;
      const v16h* bv = (const v16h*)&sB[s * 128 + q * 32];
      const v16h* cv = (const v16h*)&sC[s * 128 + q * 32];
      v16h B0 = bv[0], B1 = bv[1];
      v16h C0 = cv[0], C1 = cv[1];
#pragma unroll
      for (int j = 0; j < 16; ++j) {
        float e = __builtin_amdgcn_exp2f(dl * Ar[j]);
        hreg[j] = hreg[j] * e + du * (float)B0[j];
        yp += hreg[j] * (float)C0[j];
      }
#pragma unroll
      for (int j = 0; j < 16; ++j) {
        float e = __builtin_amdgcn_exp2f(dl * Ar[16 + j]);
        hreg[16 + j] = hreg[16 + j] * e + du * (float)B1[j];
        yp += hreg[16 + j] * (float)C1[j];
      }
      yp += __shfl_xor(yp, 1, 32);
      yp += __shfl_xor(yp, 2, 32);
      if (q == 0) ypart[((size_t)(split * Bn + b) * LPt + c * TC + s) * DI + d] = yp;
    }
  }
}

// ---------------------------------------------------------------- y = (yp0+yp1 + u*Dp) * silu(z)
__global__ void k_combine(const float* __restrict__ ypart, const _Float16* __restrict__ uth,
                          const _Float16* __restrict__ xzp, const float* __restrict__ Dp,
                          _Float16* __restrict__ ypk) {
  int i = blockIdx.x * 256 + threadIdx.x;
  if (i >= Bn * LPt * DI) return;
  int d = i % DI; int l = (i / DI) % LPt; int b = i / (DI * LPt);
  float y = ypart[(b * LPt + l) * DI + d] + ypart[((Bn + b) * LPt + l) * DI + d];
  y += (float)uth[(b * LPt + l) * DI + d] * Dp[d];
  float z = (float)xzp[pidx(b, DI + d, l, 16, T16p)];
  y *= z / (1.f + __expf(-z));
  ypk[pidx(b, d, l, 8, T16p)] = (_Float16)y;
}

// ---------------------------------------------------------------- out_proj (M=128,K=256), += into outacc
__global__ void k_outproj(const v16h* __restrict__ ypk, const v16h* __restrict__ Wop,
                          float* __restrict__ outacc, int flip) {
  int lane = threadIdx.x & 31, wv = threadIdx.x >> 5;
  int id = blockIdx.x;
  int tt4 = id % (LPt / 64); id /= (LPt / 64);
  int mt = id % 8; int b = id / 8;
  int t0 = tt4 * 64 + wv * 16;
  int tt = t0 >> 4;
  int n = lane & 15, hi = lane >> 4;
  v8f acc = {};
#pragma unroll
  for (int ks = 0; ks < 8; ++ks) {
    v16h a  = Wop[(mt * 8 + ks) * 32 + lane];
    v16h bf = ypk[(((size_t)b * 8 + ks) * T16p + tt) * 32 + lane];
    acc = WMMA_F16(a, bf, acc);
  }
  int l = t0 + n;
  int ld = flip ? (LPt - 1 - l) : l;
#pragma unroll
  for (int r = 0; r < 8; ++r) {
    int m = mt * 16 + r + hi * 8;
    outacc[(b * LPt + ld) * DM + m] += acc[r];
  }
}

// ---------------------------------------------------------------- finalize
__global__ void k_final(const float* __restrict__ outacc, const float* __restrict__ mask,
                        float* __restrict__ out) {
  int i = blockIdx.x * 256 + threadIdx.x;
  if (i >= Bn * LPt * DM) return;
  int l = (i / DM) % LPt; int b = i / (DM * LPt);
  out[i] = outacc[i] * mask[b * LPt + l];
}

__global__ void k_len(const float* __restrict__ mask, float* __restrict__ outlen) {
  __shared__ float red[256];
  int b = blockIdx.x, t = threadIdx.x;
  float s = 0.f;
  for (int i = t; i < LPt; i += 256) s += mask[b * LPt + i];
  red[t] = s; __syncthreads();
  for (int st = 128; st > 0; st >>= 1) { if (t < st) red[t] += red[t + st]; __syncthreads(); }
  if (t == 0) outlen[b] = red[0];
}

// ================================================================ host
extern "C" void kernel_launch(void* const* d_in, const int* in_sizes, int n_in,
                              void* d_out, int out_size, void* d_ws, size_t ws_size,
                              hipStream_t stream) {
  (void)in_sizes; (void)n_in; (void)out_size; (void)ws_size;
  const float* x      = (const float*)d_in[0];
  const float* mask   = (const float*)d_in[1];
  const float* b1_w1  = (const float*)d_in[2];
  const float* b1_b1  = (const float*)d_in[3];
  const float* b1_w2  = (const float*)d_in[4];
  const float* b1_b2  = (const float*)d_in[5];
  const float* b1_wd  = (const float*)d_in[6];
  const float* b1_bd  = (const float*)d_in[7];
  const float* b2_w1  = (const float*)d_in[8];
  const float* b2_b1  = (const float*)d_in[9];
  const float* b2_w2  = (const float*)d_in[10];
  const float* b2_b2  = (const float*)d_in[11];
  const float* b2_wd  = (const float*)d_in[12];
  const float* b2_bd  = (const float*)d_in[13];
  const float* ln_g   = (const float*)d_in[14];
  const float* ln_b   = (const float*)d_in[15];
  const float* in_proj= (const float*)d_in[16];
  const float* conv_w = (const float*)d_in[17];
  const float* conv_b = (const float*)d_in[18];
  const float* x_proj = (const float*)d_in[19];
  const float* dt_w   = (const float*)d_in[20];
  const float* dt_b   = (const float*)d_in[21];
  const float* A_log  = (const float*)d_in[22];
  const float* Dp     = (const float*)d_in[23];
  const float* out_pj = (const float*)d_in[24];
  float* out = (float*)d_out;

  char* base = (char*)d_ws;
  size_t off = 0;
  auto alloc = [&](size_t bytes) { size_t o = off; off += (bytes + 255) & ~(size_t)255; return o; };

  // packed weight fragment counts (halves) = tilesM * slices * 512
  const int szW1 = 16 * 2 * 512, szW2 = 16 * 16 * 512, szWD = 16 * 1 * 512;
  const int szV1 = 8 * 16 * 512, szV2 = 8 * 8 * 512, szVD = 8 * 8 * 512;
  const int szIN = 32 * 4 * 512, szXP = 33 * 8 * 512, szOP = 8 * 8 * 512;

  size_t oXP   = alloc((size_t)Bn * T16c * 512 * 2);        // x packed (ch pad 32)
  size_t oW1   = alloc((size_t)szW1 * 2);
  size_t oW2   = alloc((size_t)szW2 * 2);
  size_t oWD   = alloc((size_t)szWD * 2);
  size_t oV1   = alloc((size_t)szV1 * 2);
  size_t oV2   = alloc((size_t)szV2 * 2);
  size_t oVD   = alloc((size_t)szVD * 2);
  size_t oINW  = alloc((size_t)szIN * 2);
  size_t oXPW  = alloc((size_t)szXP * 2);
  size_t oOPW  = alloc((size_t)szOP * 2);
  size_t oAF   = alloc((size_t)DSt * DI * 4);
  size_t oBIG1 = alloc((size_t)Bn * 8 * T16c * 512 * 2);    // out1 packed | later xz packed
  size_t oBIG2 = alloc((size_t)Bn * 8 * T16c * 512 * 2);    // h1 packed   | later Bc+Cc rows
  size_t oMED1 = alloc((size_t)Bn * 4 * T16c * 512 * 2);    // o1 packed   | later u packed
  size_t oH2   = alloc((size_t)Bn * 4 * T16c * 512 * 2);    // h2 packed
  size_t oLNP  = alloc((size_t)Bn * 4 * T16p * 512 * 2);    // ln packed
  size_t oACC  = alloc((size_t)Bn * LPt * DM * 4);
  size_t oDTB  = alloc((size_t)Bn * LPt * DTR * 4);
  size_t oDEL  = alloc((size_t)Bn * LPt * DI * 4);
  size_t oUTH  = alloc((size_t)Bn * LPt * DI * 2);
  size_t oYP   = alloc((size_t)2 * Bn * LPt * DI * 4);
  size_t oYPK  = alloc((size_t)Bn * 8 * T16p * 512 * 2);    // y packed

  _Float16* XPa  = (_Float16*)(base + oXP);
  _Float16* W1P  = (_Float16*)(base + oW1);
  _Float16* W2P  = (_Float16*)(base + oW2);
  _Float16* WDP  = (_Float16*)(base + oWD);
  _Float16* V1P  = (_Float16*)(base + oV1);
  _Float16* V2P  = (_Float16*)(base + oV2);
  _Float16* VDP  = (_Float16*)(base + oVD);
  _Float16* INW  = (_Float16*)(base + oINW);
  _Float16* XPW  = (_Float16*)(base + oXPW);
  _Float16* OPW  = (_Float16*)(base + oOPW);
  float*    AF   = (float*)(base + oAF);
  _Float16* OUT1P= (_Float16*)(base + oBIG1);
  _Float16* XZP  = (_Float16*)(base + oBIG1);               // alias
  _Float16* H1P  = (_Float16*)(base + oBIG2);
  _Float16* BCH  = (_Float16*)(base + oBIG2);               // alias
  _Float16* CCH  = (_Float16*)(base + oBIG2 + (size_t)Bn * LPt * DSt * 2);
  _Float16* O1P  = (_Float16*)(base + oMED1);
  _Float16* UP   = (_Float16*)(base + oMED1);               // alias
  _Float16* H2P  = (_Float16*)(base + oH2);
  _Float16* LNP  = (_Float16*)(base + oLNP);
  float*    OACC = (float*)(base + oACC);
  float*    DTBp = (float*)(base + oDTB);
  float*    DEL  = (float*)(base + oDEL);
  _Float16* UTH  = (_Float16*)(base + oUTH);
  float*    YP   = (float*)(base + oYP);
  _Float16* YPK  = (_Float16*)(base + oYPK);

  auto cdiv = [](int a, int b) { return (a + b - 1) / b; };

  // ---- prep: pack input + all weights into WMMA fragment layouts
  k_pack_x<<<cdiv(Bn * T16c * 512, 256), 256, 0, stream>>>(x, XPa);
  k_packw<<<cdiv(szW1, 256), 256, 0, stream>>>(b1_w1, W1P, 256, 12, 32, 2, 2, szW1);
  k_packw<<<cdiv(szW2, 256), 256, 0, stream>>>(b1_w2, W2P, 256, 256, 256, 2, 16, szW2);
  k_packw<<<cdiv(szWD, 256), 256, 0, stream>>>(b1_wd, WDP, 256, 12, 32, 1, 1, szWD);
  k_packw<<<cdiv(szV1, 256), 256, 0, stream>>>(b2_w1, V1P, 128, 256, 256, 2, 16, szV1);
  k_packw<<<cdiv(szV2, 256), 256, 0, stream>>>(b2_w2, V2P, 128, 128, 128, 2, 8, szV2);
  k_packw<<<cdiv(szVD, 256), 256, 0, stream>>>(b2_wd, VDP, 128, 256, 256, 1, 8, szVD);
  k_packw<<<cdiv(szIN, 256), 256, 0, stream>>>(in_proj, INW, 512, 128, 128, 1, 4, szIN);
  k_packw<<<cdiv(szXP, 256), 256, 0, stream>>>(x_proj, XPW, 520, 256, 256, 1, 8, szXP);
  k_packw<<<cdiv(szOP, 256), 256, 0, stream>>>(out_pj, OPW, 128, 256, 256, 1, 8, szOP);
  k_negexp<<<cdiv(DSt * DI, 256), 256, 0, stream>>>(A_log, AF, DSt * DI);

  // ---- TCN
  k_tcn<32, 2, 1, 256, 0, 32><<<Bn * 16 * (Lt / 64), 128, 0, stream>>>(
      (const v16h*)XPa, (const v16h*)W1P, b1_b1, nullptr, nullptr, nullptr, OUT1P);
  k_tcn<256, 2, 1, 256, 1, 32><<<Bn * 16 * (Lt / 64), 128, 0, stream>>>(
      (const v16h*)OUT1P, (const v16h*)W2P, b1_b2,
      (const v16h*)XPa, (const v16h*)WDP, b1_bd, H1P);
  k_tcn<256, 2, 2, 128, 0, 32><<<Bn * 8 * (Lt / 64), 128, 0, stream>>>(
      (const v16h*)H1P, (const v16h*)V1P, b2_b1, nullptr, nullptr, nullptr, O1P);
  k_tcn<128, 2, 2, 128, 1, 256><<<Bn * 8 * (Lt / 64), 128, 0, stream>>>(
      (const v16h*)O1P, (const v16h*)V2P, b2_b2,
      (const v16h*)H1P, (const v16h*)VDP, b2_bd, H2P);

  // ---- pool + LN (also seeds OACC = 2*ln_out)
  k_poolln<<<Bn * LPt, 128, 0, stream>>>(H2P, ln_g, ln_b, mask, LNP, OACC);

  // ---- Mamba, forward then backward (flip)
  for (int dir = 0; dir < 2; ++dir) {
    k_inproj<<<Bn * 32 * (LPt / 64), 128, 0, stream>>>((const v16h*)LNP, (const v16h*)INW, XZP, dir);
    k_dwconv<<<cdiv(Bn * DI * LPt, 256), 256, 0, stream>>>(XZP, conv_w, conv_b, UP, UTH);
    k_xproj<<<Bn * 33 * (LPt / 64), 128, 0, stream>>>((const v16h*)UP, (const v16h*)XPW,
                                                      DTBp, BCH, CCH);
    k_delta<<<cdiv(Bn * LPt * DI, 256), 256, 0, stream>>>(DTBp, dt_w, dt_b, DEL);
    k_scan<<<Bn * 2, 1024, 0, stream>>>(AF, DEL, UTH, BCH, CCH, YP);
    k_combine<<<cdiv(Bn * LPt * DI, 256), 256, 0, stream>>>(YP, UTH, XZP, Dp, YPK);
    k_outproj<<<Bn * 8 * (LPt / 64), 128, 0, stream>>>((const v16h*)YPK, (const v16h*)OPW,
                                                       OACC, dir);
  }

  // ---- finalize
  k_final<<<cdiv(Bn * LPt * DM, 256), 256, 0, stream>>>(OACC, mask, out);
  k_len<<<Bn, 256, 0, stream>>>(mask, out + (size_t)Bn * LPt * DM);
}
